// MoeLayer_9740985827717
// MI455X (gfx1250) — matrix operand order repaired and checked by reading, not compiled, not run on hardware
//
#include <hip/hip_runtime.h>
#include <hip/hip_bf16.h>

typedef __attribute__((ext_vector_type(16))) __bf16 v16bf;
typedef __attribute__((ext_vector_type(8)))  __bf16 v8bf;
typedef __attribute__((ext_vector_type(8)))  float  v8f;

#define TOKENS 8192
#define DDIM   2048
#define ODIM   2048
#define NEXP   8
#define BM     128             // tokens per block tile
#define BN     128             // output cols per block tile
#define BK     32              // K per step (bf16 WMMA K)
#define LDA    40              // padded LDS row stride (80B: gcd(5,16)=1 -> conflict-free b128)
#define MAX_TOK_PER_EXP 8192
#define MTILES (MAX_TOK_PER_EXP / BM)   // 64

// truncation split: hi = top 16 bits of f (exact bf16 bits), lo = bf16(f - hi_f).
// f - hi_f is exact; total representation error ~2^-16 relative.
__device__ __forceinline__ void split_store(unsigned short* hi, unsigned short* lo, float f) {
  unsigned u = __builtin_bit_cast(unsigned, f);
  *hi = (unsigned short)(u >> 16);
  float hf = __builtin_bit_cast(float, u & 0xffff0000u);
  *lo = __builtin_bit_cast(unsigned short, (__bf16)(f - hf));
}

// ---------------------------------------------------------------------------
// Router: one wave32 per token -> top-2 experts, 2-way softmax, compaction.
// ---------------------------------------------------------------------------
__global__ void __launch_bounds__(256)
moe_router(const float* __restrict__ x, const float* __restrict__ gw,
           int* __restrict__ counts, int* __restrict__ tok_list,
           float* __restrict__ w_list) {
  int tok  = (blockIdx.x * blockDim.x + threadIdx.x) >> 5;
  int lane = threadIdx.x & 31;
  if (tok >= TOKENS) return;
  const float* xr = x + (size_t)tok * DDIM;
  float acc[NEXP];
#pragma unroll
  for (int e = 0; e < NEXP; ++e) acc[e] = 0.f;
  for (int d = lane; d < DDIM; d += 32) {
    float xv = xr[d];
#pragma unroll
    for (int e = 0; e < NEXP; ++e) acc[e] += xv * gw[e * DDIM + d];
  }
#pragma unroll
  for (int e = 0; e < NEXP; ++e) {
#pragma unroll
    for (int off = 16; off > 0; off >>= 1)
      acc[e] += __shfl_xor(acc[e], off, 32);
  }
  if (lane == 0) {
    int e0 = 0;
#pragma unroll
    for (int e = 1; e < NEXP; ++e)
      if (acc[e] > acc[e0]) e0 = e;            // strict > : lowest index wins ties
    int e1 = (e0 == 0) ? 1 : 0;
#pragma unroll
    for (int e = 0; e < NEXP; ++e) {
      if (e == e0 || e == e1) continue;
      if (acc[e] > acc[e1]) e1 = e;
    }
    float w1 = 1.f / (1.f + __expf(acc[e0] - acc[e1]));   // softmax over {v0,v1}
    float w0 = 1.f - w1;
    int p0 = atomicAdd(&counts[e0], 1);
    tok_list[e0 * MAX_TOK_PER_EXP + p0] = tok;
    w_list [e0 * MAX_TOK_PER_EXP + p0] = w0;
    int p1 = atomicAdd(&counts[e1], 1);
    tok_list[e1 * MAX_TOK_PER_EXP + p1] = tok;
    w_list [e1 * MAX_TOK_PER_EXP + p1] = w1;
  }
}

// ---------------------------------------------------------------------------
// Grouped expert GEMM. Block = (expert, 128-token tile, 128-col tile), 8 waves
// in a 2(M) x 4(N) grid; each wave owns 4M x 2N = eight 16x16 C tiles ->
// 24 WMMA per K-step against 24 ds_load_b128 (A frags reused across 2 N tiles,
// B frags across 4 M tiles). bf16 hi/lo split: hi*hi + hi*lo + lo*hi.
// ---------------------------------------------------------------------------
__global__ void __launch_bounds__(256, 2)
moe_expert_gemm(const float* __restrict__ x,
                const float* __restrict__ ew,   // [E][O][D]
                const float* __restrict__ eb,   // [E][O]
                const int*   __restrict__ counts,
                const int*   __restrict__ tok_list,
                const float* __restrict__ w_list,
                float* __restrict__ out) {
  int ntile = blockIdx.x;                 // 0..15
  int slot  = blockIdx.y;                 // 0..NEXP*MTILES-1
  int e     = slot >> 6;                  // / MTILES
  int mtile = slot & (MTILES - 1);
  int cnt   = counts[e];
  int m0    = mtile * BM;
  if (m0 >= cnt) return;                  // block-uniform early exit

  __shared__ __align__(16) unsigned short Ah[BM][LDA], Al[BM][LDA];
  __shared__ __align__(16) unsigned short Bh[BN][LDA], Bl[BN][LDA];
  __shared__ int   s_tok[BM];
  __shared__ float s_w[BM];

  int tid = threadIdx.x;
  if (tid < BM) {
    int m = m0 + tid;
    s_tok[tid] = (m < cnt) ? tok_list[e * MAX_TOK_PER_EXP + m]
                           : tok_list[e * MAX_TOK_PER_EXP + m0];  // pad: valid idx, w=0
    s_w[tid]   = (m < cnt) ? w_list[e * MAX_TOK_PER_EXP + m] : 0.f;
  }
  __syncthreads();

  int wid    = tid >> 5;                  // 0..7
  int lane   = tid & 31;
  int wm     = wid >> 2;                  // 0..1 : M group (64 rows each)
  int wn     = wid & 3;                   // 0..3 : N group (32 cols each)
  int n_base = ntile * BN;

  // staging assignment: every thread converts 16 A floats + 16 B floats
  int sr  = tid >> 1;                     // row 0..127 (both A and B tiles)
  int skc = (tid & 1) * 16;               // k chunk base (16 floats)
  const float* pA = x  + (size_t)s_tok[sr] * DDIM + skc;
  const float* pB = ew + (size_t)e * ODIM * DDIM + (size_t)(n_base + sr) * DDIM + skc;

  // WMMA fragment addressing (16-bit A 16x32 / B 32x16 lane layouts, wave32)
  int frow = lane & 15;
  int akb  = (lane < 16) ? 0 : 16;        // A K chunks {akb..+7, akb+8..+15}
  int bkb  = (lane < 16) ? 0 : 16;        // B K chunks {bkb..+7, bkb+8..+15}

  v8f c[4][2] = {};                        // [mt][nt]
  float4 ra[4], rb[4];

  // preload k0 = 0
#pragma unroll
  for (int q = 0; q < 4; ++q) {
    ra[q] = *(const float4*)(pA + q * 4);
    rb[q] = *(const float4*)(pB + q * 4);
  }

  for (int k0 = 0; k0 < DDIM; k0 += BK) {
    if (k0) __syncthreads();              // previous iteration's readers done
    // ---- convert + store staged registers to LDS ----
    {
      float fa[16] = {ra[0].x, ra[0].y, ra[0].z, ra[0].w, ra[1].x, ra[1].y, ra[1].z, ra[1].w,
                      ra[2].x, ra[2].y, ra[2].z, ra[2].w, ra[3].x, ra[3].y, ra[3].z, ra[3].w};
      float fb[16] = {rb[0].x, rb[0].y, rb[0].z, rb[0].w, rb[1].x, rb[1].y, rb[1].z, rb[1].w,
                      rb[2].x, rb[2].y, rb[2].z, rb[2].w, rb[3].x, rb[3].y, rb[3].z, rb[3].w};
#pragma unroll
      for (int i = 0; i < 16; ++i) {
        split_store(&Ah[sr][skc + i], &Al[sr][skc + i], fa[i]);
        split_store(&Bh[sr][skc + i], &Bl[sr][skc + i], fb[i]);
      }
    }
    __syncthreads();

    // ---- preload next K-step (global latency overlaps WMMA below) ----
    int kn = k0 + BK;
    if (kn < DDIM) {
#pragma unroll
      for (int q = 0; q < 4; ++q) {
        ra[q] = *(const float4*)(pA + kn + q * 4);
        rb[q] = *(const float4*)(pB + kn + q * 4);
      }
    }

    // ---- B fragments for this wave's 2 N sub-tiles ----
    v16bf bh[2], bl[2];
#pragma unroll
    for (int nt = 0; nt < 2; ++nt) {
      int bnr = wn * 32 + nt * 16 + frow;
      v8bf b0h = *(const v8bf*)&Bh[bnr][bkb];
      v8bf b1h = *(const v8bf*)&Bh[bnr][bkb + 8];
      v8bf b0l = *(const v8bf*)&Bl[bnr][bkb];
      v8bf b1l = *(const v8bf*)&Bl[bnr][bkb + 8];
      bh[nt] = __builtin_shufflevector(b0h, b1h, 0,1,2,3,4,5,6,7,8,9,10,11,12,13,14,15);
      bl[nt] = __builtin_shufflevector(b0l, b1l, 0,1,2,3,4,5,6,7,8,9,10,11,12,13,14,15);
    }

#pragma unroll
    for (int mt = 0; mt < 4; ++mt) {
      int am = wm * 64 + mt * 16 + frow;
      v8bf a0h = *(const v8bf*)&Ah[am][akb];
      v8bf a1h = *(const v8bf*)&Ah[am][akb + 8];
      v8bf a0l = *(const v8bf*)&Al[am][akb];
      v8bf a1l = *(const v8bf*)&Al[am][akb + 8];
      v16bf ah = __builtin_shufflevector(a0h, a1h, 0,1,2,3,4,5,6,7,8,9,10,11,12,13,14,15);
      v16bf al = __builtin_shufflevector(a0l, a1l, 0,1,2,3,4,5,6,7,8,9,10,11,12,13,14,15);
#pragma unroll
      for (int nt = 0; nt < 2; ++nt) {
        // x*w ~= hi*hi + hi*lo + lo*hi   (lo*lo ~ 2^-16 rel, dropped)
        c[mt][nt] = __builtin_amdgcn_wmma_f32_16x16x32_bf16(false, ah, false, bh[nt], (short)0, c[mt][nt], false, false);
        c[mt][nt] = __builtin_amdgcn_wmma_f32_16x16x32_bf16(false, ah, false, bl[nt], (short)0, c[mt][nt], false, false);
        c[mt][nt] = __builtin_amdgcn_wmma_f32_16x16x32_bf16(false, al, false, bh[nt], (short)0, c[mt][nt], false, false);
      }
    }
  }

  // ---- epilogue: C/D layout — VGPR r: lanes0-15 M=r, lanes16-31 M=r+8 ----
  int mbase = (lane < 16) ? 0 : 8;
#pragma unroll
  for (int nt = 0; nt < 2; ++nt) {
    int gcol  = n_base + wn * 32 + nt * 16 + (lane & 15);
    float bias = eb[e * ODIM + gcol];
#pragma unroll
    for (int mt = 0; mt < 4; ++mt) {
#pragma unroll
      for (int r = 0; r < 8; ++r) {
        int m = wm * 64 + mt * 16 + mbase + r;
        float val = s_w[m] * (c[mt][nt][r] + bias);
        unsafeAtomicAdd(&out[(size_t)s_tok[m] * ODIM + gcol], val);  // global_atomic_add_f32
      }
    }
  }
}

// ---------------------------------------------------------------------------
extern "C" void kernel_launch(void* const* d_in, const int* in_sizes, int n_in,
                              void* d_out, int out_size, void* d_ws, size_t ws_size,
                              hipStream_t stream) {
  const float* x        = (const float*)d_in[0];
  const float* gate_w   = (const float*)d_in[1];
  const float* expert_w = (const float*)d_in[2];
  const float* expert_b = (const float*)d_in[3];
  float* out = (float*)d_out;

  int*   counts   = (int*)d_ws;
  int*   tok_list = (int*)((char*)d_ws + 256);
  float* w_list   = (float*)((char*)d_ws + 256 + (size_t)NEXP * MAX_TOK_PER_EXP * sizeof(int));

  hipMemsetAsync(d_ws, 0, 256, stream);                                // counters
  hipMemsetAsync(d_out, 0, (size_t)out_size * sizeof(float), stream);  // accumulator

  moe_router<<<dim3(TOKENS / 8), 256, 0, stream>>>(x, gate_w, counts, tok_list, w_list);

  dim3 grid(ODIM / BN, NEXP * MTILES);   // (16, 512); empty tiles early-exit
  moe_expert_gemm<<<grid, 256, 0, stream>>>(x, expert_w, expert_b,
                                            counts, tok_list, w_list, out);
}
